// MambaPredictor_53343493816410
// MI455X (gfx1250) — compile-verified, matrix operand order checked by reading, not run on hardware
//
#include <hip/hip_runtime.h>
#include <hip/hip_bf16.h>
#include <math.h>

typedef __attribute__((ext_vector_type(16))) _Float16 v16h;
typedef __attribute__((ext_vector_type(8)))  float    v8f;

// ---------------------------------------------------------------------------
// device helpers
// ---------------------------------------------------------------------------
__device__ __forceinline__ float gelu_exact(float v) {
    return 0.5f * v * (1.0f + erff(v * 0.70710678118654752f));
}
__device__ __forceinline__ float softplus_f(float v) {
    return (v > 20.0f) ? v : log1pf(__expf(v));
}
__device__ __forceinline__ float silu_f(float v) {
    return v / (1.0f + __expf(-v));
}

// ---------------------------------------------------------------------------
// Generic WMMA GEMM:  C[M,N] = act(A[M,K] * W[N,K]^T + bias) + res
//   - A row-major (lda), W row-major PyTorch-style weight (ldw)
//   - one 16x16 output tile per wave (wave32), 4 waves / 128-thread block
//   - REQUIRES K % 32 == 0 and 16B-aligned rows (lda/ldw multiples of 4)
//   - edge tiles in M/N: source row index clamped (garbage only lands in
//     rows/cols that the guarded store never writes), so the K-loop is
//     completely branch-free and EXEC stays all-ones for the WMMA.
// act: 0 = none, 1 = exact GELU
// ---------------------------------------------------------------------------
__global__ void wmma_gemm_kernel(const float* __restrict__ A, int lda,
                                 const float* __restrict__ W, int ldw,
                                 const float* __restrict__ bias,
                                 const float* __restrict__ res, int ldres,
                                 float* __restrict__ C, int ldc,
                                 int M, int N, int K, int act) {
    const int lane = threadIdx.x & 31;
    const int wv   = threadIdx.x >> 5;
    const int m0   = (blockIdx.y * 4 + wv) * 16;
    const int n0   = blockIdx.x * 16;
    if (m0 >= M) return;                        // wave-uniform exit

    const int rm = lane & 15;                   // row within A frag / col within B frag
    const int kh = (lane >> 4) * 8;             // K half select (0 or 8)

    const int am = min(m0 + rm, M - 1);         // clamped: no predication needed
    const int bn = min(n0 + rm, N - 1);
    const float* arow = A + (size_t)am * lda + kh;
    const float* brow = W + (size_t)bn * ldw + kh;

    v8f acc = {};
#pragma unroll 2
    for (int k0 = 0; k0 < K; k0 += 32) {
        const float4 a0 = *reinterpret_cast<const float4*>(arow + k0);
        const float4 a1 = *reinterpret_cast<const float4*>(arow + k0 + 4);
        const float4 a2 = *reinterpret_cast<const float4*>(arow + k0 + 16);
        const float4 a3 = *reinterpret_cast<const float4*>(arow + k0 + 20);
        const float4 b0 = *reinterpret_cast<const float4*>(brow + k0);
        const float4 b1 = *reinterpret_cast<const float4*>(brow + k0 + 4);
        const float4 b2 = *reinterpret_cast<const float4*>(brow + k0 + 16);
        const float4 b3 = *reinterpret_cast<const float4*>(brow + k0 + 20);

        v16h a, b;
        a[0]=(_Float16)a0.x; a[1]=(_Float16)a0.y; a[2]=(_Float16)a0.z; a[3]=(_Float16)a0.w;
        a[4]=(_Float16)a1.x; a[5]=(_Float16)a1.y; a[6]=(_Float16)a1.z; a[7]=(_Float16)a1.w;
        a[8]=(_Float16)a2.x; a[9]=(_Float16)a2.y; a[10]=(_Float16)a2.z; a[11]=(_Float16)a2.w;
        a[12]=(_Float16)a3.x; a[13]=(_Float16)a3.y; a[14]=(_Float16)a3.z; a[15]=(_Float16)a3.w;
        b[0]=(_Float16)b0.x; b[1]=(_Float16)b0.y; b[2]=(_Float16)b0.z; b[3]=(_Float16)b0.w;
        b[4]=(_Float16)b1.x; b[5]=(_Float16)b1.y; b[6]=(_Float16)b1.z; b[7]=(_Float16)b1.w;
        b[8]=(_Float16)b2.x; b[9]=(_Float16)b2.y; b[10]=(_Float16)b2.z; b[11]=(_Float16)b2.w;
        b[12]=(_Float16)b3.x; b[13]=(_Float16)b3.y; b[14]=(_Float16)b3.z; b[15]=(_Float16)b3.w;

        acc = __builtin_amdgcn_wmma_f32_16x16x32_f16(
            /*neg_a=*/false, a, /*neg_b=*/false, b,
            /*c_mod=*/(short)0, acc, /*reuse_a=*/false, /*reuse_b=*/false);
    }

    const int nc = n0 + (lane & 15);
#pragma unroll
    for (int r = 0; r < 8; ++r) {
        const int mc = m0 + r + (lane >> 4) * 8;
        if (mc < M && nc < N) {
            float v = acc[r];
            if (bias) v += bias[nc];
            if (act == 1) v = gelu_exact(v);
            if (res)  v += res[(size_t)mc * ldres + nc];
            C[(size_t)mc * ldc + nc] = v;
        }
    }
}

// ---------------------------------------------------------------------------
// zero-padded pack: dst[rows, ldd] = src[rows, 0:cols] (lds leading dim), pad 0
// ---------------------------------------------------------------------------
__global__ void pack_pad_kernel(const float* __restrict__ src, int lds, int cols,
                                float* __restrict__ dst, int ldd, int rows) {
    const int i = blockIdx.x * blockDim.x + threadIdx.x;
    const int total = rows * ldd;
    if (i >= total) return;
    const int c = i % ldd;
    const int r = i / ldd;
    dst[i] = (c < cols) ? src[(size_t)r * lds + c] : 0.0f;
}

// ---------------------------------------------------------------------------
// LayerNorm over last dim D (wave per row), optional exact GELU
// ---------------------------------------------------------------------------
__global__ void ln_kernel(const float* __restrict__ X,
                          const float* __restrict__ g,
                          const float* __restrict__ b,
                          float* __restrict__ Y, int rows, int D, int dogelu) {
    const int wv = threadIdx.x >> 5, lane = threadIdx.x & 31;
    const int row = blockIdx.x * (blockDim.x >> 5) + wv;
    if (row >= rows) return;
    const float* x = X + (size_t)row * D;
    float s = 0.0f, ss = 0.0f;
    for (int i = lane; i < D; i += 32) { const float v = x[i]; s += v; ss += v * v; }
#pragma unroll
    for (int off = 16; off >= 1; off >>= 1) {
        s  += __shfl_xor(s,  off, 32);
        ss += __shfl_xor(ss, off, 32);
    }
    const float mean = s / (float)D;
    const float var  = ss / (float)D - mean * mean;
    const float inv  = rsqrtf(var + 1e-5f);
    float* y = Y + (size_t)row * D;
    for (int i = lane; i < D; i += 32) {
        float v = (x[i] - mean) * inv * g[i] + b[i];
        if (dogelu) v = gelu_exact(v);
        y[i] = v;
    }
}

// ---------------------------------------------------------------------------
// Causal depthwise conv (kernel 4) + SiLU on the u half of the in-proj output
// ---------------------------------------------------------------------------
__global__ void conv_silu_kernel(const float* __restrict__ uz,
                                 const float* __restrict__ cw,
                                 const float* __restrict__ cb,
                                 float* __restrict__ u,
                                 int Bv, int Lv, int DIv, int lduz) {
    const int idx = blockIdx.x * blockDim.x + threadIdx.x;
    const int total = Bv * Lv * DIv;
    if (idx >= total) return;
    const int d  = idx % DIv;
    const int t  = (idx / DIv) % Lv;
    const int bb = idx / (DIv * Lv);
    float acc = cb[d];
#pragma unroll
    for (int j = 0; j < 4; ++j) {
        const int tt = t - 3 + j;
        if (tt >= 0)
            acc += uz[((size_t)(bb * Lv + tt)) * lduz + d] * cw[d * 4 + j];
    }
    u[(size_t)idx] = silu_f(acc);
}

// ---------------------------------------------------------------------------
// dt = softplus(dt_raw + dt_b), in place
// ---------------------------------------------------------------------------
__global__ void softplus_bias_kernel(float* __restrict__ dt,
                                     const float* __restrict__ dtb,
                                     int total, int DIv) {
    const int i = blockIdx.x * blockDim.x + threadIdx.x;
    if (i >= total) return;
    dt[i] = softplus_f(dt[i] + dtb[i % DIv]);
}

// ---------------------------------------------------------------------------
// Selective scan: one thread per (batch, channel), NS states in registers.
// xd rows hold [dt_in(16) | B(NS) | C(NS)] with leading dim ldxd.
// ---------------------------------------------------------------------------
template <int NS>
__global__ void scan_kernel(const float* __restrict__ dt,
                            const float* __restrict__ u,
                            const float* __restrict__ xd, int ldxd,
                            const float* __restrict__ Alog,
                            const float* __restrict__ Dp,
                            float* __restrict__ y,
                            int Bv, int Lv, int DIv) {
    const int idx = blockIdx.x * blockDim.x + threadIdx.x;
    if (idx >= Bv * DIv) return;
    const int d  = idx % DIv;
    const int bb = idx / DIv;
    float a[NS], h[NS];
#pragma unroll
    for (int n = 0; n < NS; ++n) {
        a[n] = -__expf(Alog[d * NS + n]);
        h[n] = 0.0f;
    }
    const float Dd = Dp[d];
    for (int t = 0; t < Lv; ++t) {
        const size_t r = (size_t)(bb * Lv + t);
        const float dtv = dt[r * DIv + d];
        const float uv  = u[r * DIv + d];
        const float* xr = xd + r * ldxd + 16;   // B at DTR, C at DTR+NS
        float yt = 0.0f;
#pragma unroll
        for (int n = 0; n < NS; ++n) {
            const float dA = __expf(dtv * a[n]);
            h[n] = dA * h[n] + dtv * xr[n] * uv;
            yt  += h[n] * xr[NS + n];
        }
        y[r * DIv + d] = yt + uv * Dd;
    }
}

// ---------------------------------------------------------------------------
// y *= silu(z) with z = uz[row, DI + d]  (in place on y)
// ---------------------------------------------------------------------------
__global__ void gate_kernel(float* __restrict__ y, const float* __restrict__ uz,
                            int total, int DIv, int lduz) {
    const int i = blockIdx.x * blockDim.x + threadIdx.x;
    if (i >= total) return;
    const int d = i % DIv;
    const size_t row = (size_t)(i / DIv);
    const float z = uz[row * lduz + DIv + d];
    y[i] = y[i] * silu_f(z);
}

// ---------------------------------------------------------------------------
// hs = h[:, L-s:, :] compacted to (B*s, D)
// ---------------------------------------------------------------------------
__global__ void suffix_kernel(const float* __restrict__ h, float* __restrict__ hs,
                              int Bv, int Lv, int s, int D) {
    const int i = blockIdx.x * blockDim.x + threadIdx.x;
    const int total = Bv * s * D;
    if (i >= total) return;
    const int j  = i % D;
    const int r  = i / D;
    const int ii = r % s;
    const int bb = r / s;
    hs[i] = h[((size_t)(bb * Lv + (Lv - s) + ii)) * D + j];
}

// ---------------------------------------------------------------------------
// stack[b, si, :] = mean over s tokens of hs2
// ---------------------------------------------------------------------------
__global__ void pool_kernel(const float* __restrict__ hs2, float* __restrict__ stack,
                            int Bv, int s, int si, int D) {
    const int i = blockIdx.x * blockDim.x + threadIdx.x;
    if (i >= Bv * D) return;
    const int j  = i % D;
    const int bb = i / D;
    float acc = 0.0f;
    for (int t = 0; t < s; ++t) acc += hs2[((size_t)(bb * s + t)) * D + j];
    stack[((size_t)(bb * 3 + si)) * D + j] = acc / (float)s;
}

// ---------------------------------------------------------------------------
// 3-token multi-head attention (softmax over S keys), one thread per (b,h,q)
// qkv rows: (b*S + s) x (3E) laid out [q | k | v]
// ---------------------------------------------------------------------------
__global__ void mha3_kernel(const float* __restrict__ qkv, float* __restrict__ o,
                            int Bv, int S, int E, int H) {
    const int i = blockIdx.x * blockDim.x + threadIdx.x;
    if (i >= Bv * H * S) return;
    const int q  = i % S;
    const int hh = (i / S) % H;
    const int bb = i / (S * H);
    const int hd = E / H;
    const float* qv = qkv + ((size_t)(bb * S + q)) * (3 * E) + hh * hd;
    float sc[4];
    float mx = -1e30f;
    for (int t = 0; t < S; ++t) {
        const float* kv = qkv + ((size_t)(bb * S + t)) * (3 * E) + E + hh * hd;
        float dot = 0.0f;
        for (int d = 0; d < hd; ++d) dot += qv[d] * kv[d];
        sc[t] = dot * rsqrtf((float)hd);
        mx = fmaxf(mx, sc[t]);
    }
    float den = 0.0f;
    for (int t = 0; t < S; ++t) { sc[t] = __expf(sc[t] - mx); den += sc[t]; }
    float* orow = o + ((size_t)(bb * S + q)) * E + hh * hd;
    for (int d = 0; d < hd; ++d) {
        float acc = 0.0f;
        for (int t = 0; t < S; ++t)
            acc += sc[t] * qkv[((size_t)(bb * S + t)) * (3 * E) + 2 * E + hh * hd + d];
        orow[d] = acc / den;
    }
}

// final = fused + att.mean(over S tokens)
__global__ void final_kernel(const float* __restrict__ fused,
                             const float* __restrict__ att,
                             float* __restrict__ fin, int Bv, int S, int D) {
    const int i = blockIdx.x * blockDim.x + threadIdx.x;
    if (i >= Bv * D) return;
    const int j  = i % D;
    const int bb = i / D;
    float acc = 0.0f;
    for (int t = 0; t < S; ++t) acc += att[((size_t)(bb * S + t)) * D + j];
    fin[i] = fused[i] + acc / (float)S;
}

// ---------------------------------------------------------------------------
// Evidential/TTF/emb head post-processing + final concat -> d_out (B x 162)
// ---------------------------------------------------------------------------
__global__ void heads_kernel(const float* __restrict__ evraw,  // B x 7
                             const float* __restrict__ ttf,    // B x 12
                             const float* __restrict__ emb,    // B x 128
                             float* __restrict__ out, int Bv) {
    const int bb = blockIdx.x * blockDim.x + threadIdx.x;
    if (bb >= Bv) return;
    float ev[7], alpha[7];
    float Ssum = 0.0f;
#pragma unroll
    for (int c = 0; c < 7; ++c) {
        const float v = softplus_f(evraw[bb * 7 + c]);
        ev[c] = v;
        alpha[c] = v + 1.0f;
        Ssum += alpha[c];
    }
    float* o = out + (size_t)bb * 162;
#pragma unroll
    for (int c = 0; c < 7; ++c) o[c]      = alpha[c] / Ssum;   // probs
#pragma unroll
    for (int c = 0; c < 7; ++c) o[7 + c]  = ev[c];
#pragma unroll
    for (int c = 0; c < 7; ++c) o[14 + c] = alpha[c];
    o[21] = 7.0f / Ssum;                                        // unc
#pragma unroll
    for (int j = 0; j < 6; ++j) o[22 + j] = softplus_f(ttf[bb * 12 + j]);        // tm
#pragma unroll
    for (int j = 0; j < 6; ++j) o[28 + j] = __expf(0.5f * ttf[bb * 12 + 6 + j]); // ts
    for (int j = 0; j < 128; ++j) o[34 + j] = emb[bb * 128 + j];
}

// ---------------------------------------------------------------------------
// host-side helpers
// ---------------------------------------------------------------------------
static inline void gemm(hipStream_t st, const float* A, int lda,
                        const float* W, int ldw, const float* bias,
                        const float* res, int ldres, float* C, int ldc,
                        int M, int N, int K, int act) {
    dim3 grid((N + 15) / 16, (M + 63) / 64);
    wmma_gemm_kernel<<<grid, 128, 0, st>>>(A, lda, W, ldw, bias, res, ldres,
                                           C, ldc, M, N, K, act);
}

static void run_mamba(hipStream_t st, const float* hin, float* hout,
                      int Bv, int Lv, int ds, int ldxd,
                      const float* ln_g, const float* ln_b, const float* in_w,
                      const float* cw, const float* cb, const float* xp_w,
                      const float* dt_w, const float* dt_b, const float* Alog,
                      const float* Dp, const float* out_w,
                      float* xn, float* uz, float* u, float* xd, float* xdt,
                      float* dtwp, float* dt, float* y) {
    const int M = Bv * Lv, DI = 512, DM = 256;
    ln_kernel<<<(M + 7) / 8, 256, 0, st>>>(hin, ln_g, ln_b, xn, M, DM, 0);
    gemm(st, xn, DM, in_w, DM, nullptr, nullptr, 0, uz, 2 * DI, M, 2 * DI, DM, 0);
    const int tot = M * DI;
    conv_silu_kernel<<<(tot + 255) / 256, 256, 0, st>>>(uz, cw, cb, u, Bv, Lv, DI, 2 * DI);
    gemm(st, u, DI, xp_w, DI, nullptr, nullptr, 0, xd, ldxd, M, ldxd, DI, 0);
    // zero-pad K=16 operands of the dt projection up to K=32
    pack_pad_kernel<<<(M * 32 + 255) / 256, 256, 0, st>>>(xd, ldxd, 16, xdt, 32, M);
    pack_pad_kernel<<<(DI * 32 + 255) / 256, 256, 0, st>>>(dt_w, 16, 16, dtwp, 32, DI);
    gemm(st, xdt, 32, dtwp, 32, nullptr, nullptr, 0, dt, DI, M, DI, 32, 0);
    softplus_bias_kernel<<<(tot + 255) / 256, 256, 0, st>>>(dt, dt_b, tot, DI);
    const int sth = Bv * DI;
    if (ds == 16)
        scan_kernel<16><<<(sth + 255) / 256, 256, 0, st>>>(dt, u, xd, ldxd, Alog, Dp, y, Bv, Lv, DI);
    else
        scan_kernel<8><<<(sth + 255) / 256, 256, 0, st>>>(dt, u, xd, ldxd, Alog, Dp, y, Bv, Lv, DI);
    gate_kernel<<<(tot + 255) / 256, 256, 0, st>>>(y, uz, tot, DI, 2 * DI);
    gemm(st, y, DI, out_w, DI, nullptr, hin, DM, hout, DM, M, DM, DI, 0);
}

// ---------------------------------------------------------------------------
extern "C" void kernel_launch(void* const* d_in, const int* in_sizes, int n_in,
                              void* d_out, int out_size, void* d_ws, size_t ws_size,
                              hipStream_t stream) {
    (void)in_sizes; (void)n_in; (void)out_size; (void)ws_size;
    const float* x        = (const float*)d_in[0];
    const float* in_w     = (const float*)d_in[1];
    const float* in_b     = (const float*)d_in[2];
    const float* in_ln_g  = (const float*)d_in[3];
    const float* in_ln_b  = (const float*)d_in[4];
    const float* mb_ln_g  = (const float*)d_in[5];
    const float* mb_ln_b  = (const float*)d_in[6];
    const float* mb_in_w  = (const float*)d_in[7];
    const float* mb_conv_w= (const float*)d_in[8];
    const float* mb_conv_b= (const float*)d_in[9];
    const float* mb_xp_w  = (const float*)d_in[10];
    const float* mb_dt_w  = (const float*)d_in[11];
    const float* mb_dt_b  = (const float*)d_in[12];
    const float* mb_Alog  = (const float*)d_in[13];
    const float* mb_D     = (const float*)d_in[14];
    const float* mb_out_w = (const float*)d_in[15];
    const float* sc_ln_g  = (const float*)d_in[16];
    const float* sc_ln_b  = (const float*)d_in[17];
    const float* sc_in_w  = (const float*)d_in[18];
    const float* sc_conv_w= (const float*)d_in[19];
    const float* sc_conv_b= (const float*)d_in[20];
    const float* sc_xp_w  = (const float*)d_in[21];
    const float* sc_dt_w  = (const float*)d_in[22];
    const float* sc_dt_b  = (const float*)d_in[23];
    const float* sc_Alog  = (const float*)d_in[24];
    const float* sc_D     = (const float*)d_in[25];
    const float* sc_out_w = (const float*)d_in[26];
    const float* fus_qkv_w= (const float*)d_in[27];
    const float* fus_qkv_b= (const float*)d_in[28];
    const float* fus_out_w= (const float*)d_in[29];
    const float* fus_out_b= (const float*)d_in[30];
    const float* ms_out_w = (const float*)d_in[31];
    const float* ms_out_b = (const float*)d_in[32];
    const float* sa_ln1_g = (const float*)d_in[33];
    const float* sa_ln1_b = (const float*)d_in[34];
    const float* sa_ln2_g = (const float*)d_in[35];
    const float* sa_ln2_b = (const float*)d_in[36];
    const float* sa_qkv_w = (const float*)d_in[37];
    const float* sa_qkv_b = (const float*)d_in[38];
    const float* sa_out_w = (const float*)d_in[39];
    const float* sa_out_b = (const float*)d_in[40];
    const float* sa_ff1_w = (const float*)d_in[41];
    const float* sa_ff1_b = (const float*)d_in[42];
    const float* sa_ff2_w = (const float*)d_in[43];
    const float* sa_ff2_b = (const float*)d_in[44];
    const float* ev1_w    = (const float*)d_in[45];
    const float* ev1_b    = (const float*)d_in[46];
    const float* ev2_w    = (const float*)d_in[47];
    const float* ev2_b    = (const float*)d_in[48];
    const float* ttf1_w   = (const float*)d_in[49];
    const float* ttf1_b   = (const float*)d_in[50];
    const float* ttf2_w   = (const float*)d_in[51];
    const float* ttf2_b   = (const float*)d_in[52];
    const float* emb_w    = (const float*)d_in[53];
    const float* emb_b    = (const float*)d_in[54];

    const int B = 4, L = 1024, DM = 256, DI = 512;
    const size_t ML = (size_t)B * L;             // 4096

    // workspace layout (floats)
    float* W = (float*)d_ws;
    size_t off = 0;
    auto alloc = [&](size_t n) { float* p = W + off; off += n; return p; };
    float* h    = alloc(ML * DM);
    float* h2   = alloc(ML * DM);
    float* xn   = alloc(ML * DM);
    float* uz   = alloc(ML * 2 * DI);
    float* u    = alloc(ML * DI);
    float* xd   = alloc(ML * 48);
    float* xdt  = alloc(ML * 32);
    float* dtwp = alloc((size_t)DI * 32);
    float* dt   = alloc(ML * DI);
    float* y    = alloc(ML * DI);
    float* hs   = alloc(200 * DM);
    float* hs2  = alloc(200 * DM);
    float* stack= alloc((size_t)B * 3 * DM);
    float* qkvb = alloc((size_t)12 * 768);
    float* mo   = alloc((size_t)12 * DM);
    float* mf   = alloc((size_t)12 * DM);      // == (B, 768)
    float* fusedv = alloc((size_t)B * DM);
    float* lnb  = alloc((size_t)12 * DM);
    float* s2   = alloc((size_t)12 * DM);
    float* ffb  = alloc((size_t)12 * 1024);
    float* attb = alloc((size_t)12 * DM);
    float* finv = alloc((size_t)B * DM);
    float* evh  = alloc((size_t)B * DM);
    float* evraw= alloc((size_t)B * 8);
    float* ttfh = alloc((size_t)B * 128);
    float* ttfb = alloc((size_t)B * 16);
    float* embv = alloc((size_t)B * 128);

    // ---- input projection + LN + GELU -> h
    gemm(stream, x, 64, in_w, 64, in_b, nullptr, 0, xn, DM, (int)ML, DM, 64, 0);
    ln_kernel<<<((int)ML + 7) / 8, 256, 0, stream>>>(xn, in_ln_g, in_ln_b, h, (int)ML, DM, 1);

    // ---- 4 mamba layers (ds=16, xd row = 48)
    float* cur = h; float* nxt = h2;
    for (int i = 0; i < 4; ++i) {
        run_mamba(stream, cur, nxt, B, L, 16, 48,
                  mb_ln_g + i * DM, mb_ln_b + i * DM,
                  mb_in_w + (size_t)i * 2 * DI * DM,
                  mb_conv_w + (size_t)i * DI * 4, mb_conv_b + (size_t)i * DI,
                  mb_xp_w + (size_t)i * 48 * DI,
                  mb_dt_w + (size_t)i * DI * 16, mb_dt_b + (size_t)i * DI,
                  mb_Alog + (size_t)i * DI * 16, mb_D + (size_t)i * DI,
                  mb_out_w + (size_t)i * DM * DI,
                  xn, uz, u, xd, xdt, dtwp, dt, y);
        float* t = cur; cur = nxt; nxt = t;
    }
    // after 4 swaps result is back in h (cur == h)

    // ---- 3 scale branches (ds=8, xd row = 32) -> stack (B,3,DM)
    const int scales[3] = {5, 20, 50};
    for (int si = 0; si < 3; ++si) {
        const int s = scales[si];
        const int tot = B * s * DM;
        suffix_kernel<<<(tot + 255) / 256, 256, 0, stream>>>(cur, hs, B, L, s, DM);
        run_mamba(stream, hs, hs2, B, s, 8, 32,
                  sc_ln_g + si * DM, sc_ln_b + si * DM,
                  sc_in_w + (size_t)si * 2 * DI * DM,
                  sc_conv_w + (size_t)si * DI * 4, sc_conv_b + (size_t)si * DI,
                  sc_xp_w + (size_t)si * 32 * DI,
                  sc_dt_w + (size_t)si * DI * 16, sc_dt_b + (size_t)si * DI,
                  sc_Alog + (size_t)si * DI * 8, sc_D + (size_t)si * DI,
                  sc_out_w + (size_t)si * DM * DI,
                  xn, uz, u, xd, xdt, dtwp, dt, y);
        pool_kernel<<<(B * DM + 255) / 256, 256, 0, stream>>>(hs2, stack, B, s, si, DM);
    }

    // ---- fused branch: MHA over 3 tokens, flatten, ms_out
    gemm(stream, stack, DM, fus_qkv_w, DM, fus_qkv_b, nullptr, 0, qkvb, 768, 12, 768, DM, 0);
    mha3_kernel<<<1, 64, 0, stream>>>(qkvb, mo, B, 3, DM, 4);
    gemm(stream, mo, DM, fus_out_w, DM, fus_out_b, nullptr, 0, mf, DM, 12, DM, DM, 0);
    gemm(stream, mf, 768, ms_out_w, 768, ms_out_b, nullptr, 0, fusedv, DM, B, DM, 768, 0);

    // ---- self-attention block on stack
    ln_kernel<<<2, 256, 0, stream>>>(stack, sa_ln1_g, sa_ln1_b, lnb, 12, DM, 0);
    gemm(stream, lnb, DM, sa_qkv_w, DM, sa_qkv_b, nullptr, 0, qkvb, 768, 12, 768, DM, 0);
    mha3_kernel<<<1, 64, 0, stream>>>(qkvb, mo, B, 3, DM, 4);
    gemm(stream, mo, DM, sa_out_w, DM, sa_out_b, stack, DM, s2, DM, 12, DM, DM, 0);
    ln_kernel<<<2, 256, 0, stream>>>(s2, sa_ln2_g, sa_ln2_b, lnb, 12, DM, 0);
    gemm(stream, lnb, DM, sa_ff1_w, DM, sa_ff1_b, nullptr, 0, ffb, 1024, 12, 1024, DM, 1);
    gemm(stream, ffb, 1024, sa_ff2_w, 1024, sa_ff2_b, s2, DM, attb, DM, 12, DM, 1024, 0);
    final_kernel<<<(B * DM + 255) / 256, 256, 0, stream>>>(fusedv, attb, finv, B, 3, DM);

    // ---- heads
    gemm(stream, finv, DM, ev1_w, DM, ev1_b, nullptr, 0, evh, DM, B, DM, DM, 1);
    gemm(stream, evh, DM, ev2_w, DM, ev2_b, nullptr, 0, evraw, 7, B, 7, DM, 0);
    gemm(stream, finv, DM, ttf1_w, DM, ttf1_b, nullptr, 0, ttfh, 128, B, 128, DM, 1);
    gemm(stream, ttfh, 128, ttf2_w, 128, ttf2_b, nullptr, 0, ttfb, 12, B, 12, 128, 0);
    gemm(stream, finv, DM, emb_w, DM, emb_b, nullptr, 0, embv, 128, B, 128, DM, 0);
    heads_kernel<<<1, 32, 0, stream>>>(evraw, ttfb, embv, (float*)d_out, B);
}
